// CCLayer_86277303042195
// MI455X (gfx1250) — matrix operand order repaired and checked by reading, not compiled, not run on hardware
//
#include <hip/hip_runtime.h>
#include <hip/hip_bf16.h>
#include <math.h>

// Problem constants (from reference)
#define DD     64
#define KK     8
#define NCOLS  262144
#define GAMMA  (-0.5f)

// W has 528 rows: rows [0,512) = A_k stacked (r = k*64 + d), rows [512,520) = U^T, [520,528) zero
#define WROWS  528
#define CTILES 8           // 32-column tiles per block -> 256 columns per block

// LDS strides chosen for bank-conflict-free access
#define XT_S   72          // halfs, f16 transposed X tile [n][e]
#define P2_S   33          // floats, partial squared-sum buffer row stride (32 cols + pad)
#define UTX_S  36          // floats, uTX buffer row stride
#define GK_S   9           // floats, pi_l/gk_l row stride (8 + pad)

typedef _Float16 v16h __attribute__((ext_vector_type(16)));
typedef _Float16 v8h  __attribute__((ext_vector_type(8)));
typedef float    v8f  __attribute__((ext_vector_type(8)));

static __device__ __forceinline__ v16h concat8(v8h a, v8h b) {
  return __builtin_shufflevector(a, b, 0,1,2,3,4,5,6,7,8,9,10,11,12,13,14,15);
}

// Horizontal sum of v8f via pairwise tree: packed ops on adjacent register pairs.
static __device__ __forceinline__ float hsum8(v8f s) {
  const float a = (s[0] + s[1]) + (s[2] + s[3]);
  const float b = (s[4] + s[5]) + (s[6] + s[7]);
  return a + b;
}

// ---------------- Prologue: build W (f16) and c (f32) in workspace ----------------
__global__ __launch_bounds__(256) void cc_prep(const float* __restrict__ NA,
                                               const float* __restrict__ Nmu,
                                               const float* __restrict__ U,
                                               _Float16* __restrict__ W,
                                               float* __restrict__ cvec) {
  const int t = threadIdx.x;
  // rows 0..511: W[r][e] = NA[d][e][k], c[r] = sum_e NA[d][e][k]*Nmu[e][k]
  for (int r = t; r < 512; r += 256) {
    const int d = r & 63, k = r >> 6;
    float acc = 0.0f;
    for (int e = 0; e < DD; ++e) {
      float a = NA[(d * DD + e) * KK + k];
      W[r * DD + e] = (_Float16)a;
      acc += a * Nmu[e * KK + k];
    }
    cvec[r] = acc;
  }
  // rows 512..527: U^T (8 rows) then zeros
  for (int r = 512 + t; r < WROWS; r += 256) {
    const int k = r - 512;
    for (int e = 0; e < DD; ++e)
      W[r * DD + e] = (k < KK) ? (_Float16)U[e * KK + k] : (_Float16)0.0f;
  }
}

// ---------------- Main kernel ----------------
__global__ __launch_bounds__(256) void cc_main(const float* __restrict__ XPi,
                                               const float* __restrict__ U,
                                               const float* __restrict__ alpha,
                                               const _Float16* __restrict__ W,
                                               const float* __restrict__ cvec,
                                               float* __restrict__ out) {
  __shared__ _Float16 xt[32 * XT_S];     // f16 X tile, transposed [n][e]
  __shared__ float    xf[64 * 32];       // f32 X tile, natural [e][n]
  __shared__ float    part2[64 * P2_S];  // partial squared sums [tile*2+half][n]
  __shared__ float    utx[16 * UTX_S];   // raw uTX rows (tile 32) [j + 8*half][n]
  __shared__ float    u_lds[512];
  __shared__ float    alpha_lds[8];
  __shared__ float    pi_l[32 * GK_S];   // [n][k]
  __shared__ float    gk_l[32 * GK_S];   // [n][k]

  const int t    = threadIdx.x;
  const int lane = t & 31;
  const int wave = t >> 5;                            // 0..7
  const int wuni = __builtin_amdgcn_readfirstlane(wave);
  const int ntiles = (wuni == 0) ? 5 : 4;             // wave 0 also owns the U^T tile (tile 32)
  const int half = lane >> 4;                         // 0/1
  const int n16  = lane & 15;

  // Stage small constants into LDS
  for (int i = t; i < 512; i += 256) u_lds[i] = U[i];
  if (t < 8) alpha_lds[t] = alpha[t];

  // ---- Load persistent A fragments (16-bit A layout: lane = M, K chunks 8-wide) ----
  const int ac0 = half * 16;  // 0 or 16
  v16h afr[5][2];
#pragma unroll
  for (int i = 0; i < 5; ++i) {
    if (i < ntiles) {                 // scalar-uniform branch (EXEC stays full)
      const int tile = wuni + 8 * i;  // 0..32
      const _Float16* wp = W + (tile * 16 + n16) * DD;
      v8h a0 = *(const v8h*)(wp + ac0);
      v8h a1 = *(const v8h*)(wp + ac0 + 8);
      afr[i][0] = concat8(a0, a1);    // K 0..31
      v8h a2 = *(const v8h*)(wp + ac0 + 32);
      v8h a3 = *(const v8h*)(wp + ac0 + 40);
      afr[i][1] = concat8(a2, a3);    // K 32..63
    }
  }

  // ---- Preload c slices (column-invariant): creg[i][j] = c[tile*16 + 8*half + j] ----
  v8f creg[4];
#pragma unroll
  for (int i = 0; i < 4; ++i) {
    const int rowb = (wuni + 8 * i) * 16 + 8 * half;
    creg[i] = *(const v8f*)(cvec + rowb);   // 32B-aligned
  }

  const int colblock0 = blockIdx.x * (32 * CTILES);

  for (int ct = 0; ct < CTILES; ++ct) {
    const int col = colblock0 + ct * 32;

    // ---- Load X tile (64 rows x 32 cols f32), store f16 transposed + f32 natural ----
    {
      const int e0 = t >> 2;            // 0..63
      const int n0 = (t & 3) * 8;       // 0,8,16,24
      const float4 va = *(const float4*)(XPi + e0 * NCOLS + col + n0);
      const float4 vb = *(const float4*)(XPi + e0 * NCOLS + col + n0 + 4);
      xt[(n0 + 0) * XT_S + e0] = (_Float16)va.x;
      xt[(n0 + 1) * XT_S + e0] = (_Float16)va.y;
      xt[(n0 + 2) * XT_S + e0] = (_Float16)va.z;
      xt[(n0 + 3) * XT_S + e0] = (_Float16)va.w;
      xt[(n0 + 4) * XT_S + e0] = (_Float16)vb.x;
      xt[(n0 + 5) * XT_S + e0] = (_Float16)vb.y;
      xt[(n0 + 6) * XT_S + e0] = (_Float16)vb.z;
      xt[(n0 + 7) * XT_S + e0] = (_Float16)vb.w;
      *(float4*)&xf[e0 * 32 + n0]     = va;   // ds_store_b128
      *(float4*)&xf[e0 * 32 + n0 + 4] = vb;   // ds_store_b128
    }
    __syncthreads();

    // ---- B fragments for both 16-column halves ----
    v16h bA0, bA1, bB0, bB1;
    {
      const _Float16* bpA = &xt[n16 * XT_S + half * 16];
      const _Float16* bpB = &xt[(n16 + 16) * XT_S + half * 16];
      bA0 = concat8(*(const v8h*)(bpA),      *(const v8h*)(bpA + 8));    // cols 0-15, K 0..31
      bA1 = concat8(*(const v8h*)(bpA + 32), *(const v8h*)(bpA + 40));   // cols 0-15, K 32..63
      bB0 = concat8(*(const v8h*)(bpB),      *(const v8h*)(bpB + 8));    // cols 16-31, K 0..31
      bB1 = concat8(*(const v8h*)(bpB + 32), *(const v8h*)(bpB + 40));   // cols 16-31, K 32..63
    }

    // ---- GEMM (two independent column halves) with fused (v-c)^2 partials ----
#pragma unroll
    for (int i = 0; i < 5; ++i) {
      if (i < ntiles) {                 // scalar-uniform
        const int tile = wuni + 8 * i;
        v8f accA = {}, accB = {};
        accA = __builtin_amdgcn_wmma_f32_16x16x32_f16(false, afr[i][0], false, bA0,
                                                      (short)0, accA, false, false);
        accB = __builtin_amdgcn_wmma_f32_16x16x32_f16(false, afr[i][0], false, bB0,
                                                      (short)0, accB, false, false);
        accA = __builtin_amdgcn_wmma_f32_16x16x32_f16(false, afr[i][1], false, bA1,
                                                      (short)0, accA, false, false);
        accB = __builtin_amdgcn_wmma_f32_16x16x32_f16(false, afr[i][1], false, bB1,
                                                      (short)0, accB, false, false);
        if (i < 4) {
          // Whole-vector (v-c)^2: packed ops on adjacent pairs of one acc, no shuffles
          v8f dA = accA - creg[i];
          v8f dB = accB - creg[i];
          dA = dA * dA;
          dB = dB * dB;
          part2[(tile * 2 + half) * P2_S + n16]      = hsum8(dA);
          part2[(tile * 2 + half) * P2_S + n16 + 16] = hsum8(dB);
        } else {
          // tile 32: raw uTX rows (512..519 useful; 520..527 zeros)
#pragma unroll
          for (int j = 0; j < 8; ++j) {
            utx[(j + 8 * half) * UTX_S + n16]      = accA[j];
            utx[(j + 8 * half) * UTX_S + n16 + 16] = accB[j];
          }
        }
      }
    }
    __syncthreads();

    // ---- Softmax + gate, one thread per column (full wave of 32) ----
    if (t < 32) {
      const int n = t;
      float lg[8];
      float mx = -1e30f;
#pragma unroll
      for (int k = 0; k < KK; ++k) {
        float s = 0.0f;
#pragma unroll
        for (int j = 0; j < 8; ++j) s += part2[(k * 8 + j) * P2_S + n];
        lg[k] = GAMMA * s;
        mx = fmaxf(mx, lg[k]);
      }
      float ex[8], sum = 0.0f;
#pragma unroll
      for (int k = 0; k < KK; ++k) { ex[k] = __expf(lg[k] - mx); sum += ex[k]; }
      const float inv = 1.0f / sum;
#pragma unroll
      for (int k = 0; k < KK; ++k) {
        const float Pi = ex[k] * inv;
        pi_l[n * GK_S + k] = Pi;
        const float tk = utx[k * UTX_S + n];         // uTX[n][k]
        gk_l[n * GK_S + k] = Pi * (tk - alpha_lds[k]);
      }
    }
    __syncthreads();

    // ---- Outputs: X_out = x - U * gk ; rows 64..71 = Pi ----
    {
      const int n = t & 31, g = t >> 5;   // g == wave, uniform per wave
      float gk[8];
#pragma unroll
      for (int k = 0; k < KK; ++k) gk[k] = gk_l[n * GK_S + k];
#pragma unroll
      for (int ii = 0; ii < 8; ++ii) {
        const int d = g + 8 * ii;
        float acc = xf[d * 32 + n];
#pragma unroll
        for (int k = 0; k < KK; ++k) acc -= u_lds[d * 8 + k] * gk[k];
        out[d * NCOLS + col + n] = acc;   // coalesced: 32 consecutive lanes -> 128B
      }
      out[(64 + g) * NCOLS + col + n] = pi_l[n * GK_S + g];
    }
    __syncthreads();
  }
}

// ---------------- Host launcher ----------------
extern "C" void kernel_launch(void* const* d_in, const int* in_sizes, int n_in,
                              void* d_out, int out_size, void* d_ws, size_t ws_size,
                              hipStream_t stream) {
  const float* XPi   = (const float*)d_in[0];
  const float* U     = (const float*)d_in[1];
  const float* alpha = (const float*)d_in[2];
  const float* NA    = (const float*)d_in[3];
  const float* Nmu   = (const float*)d_in[4];
  float* out = (float*)d_out;

  _Float16* W    = (_Float16*)d_ws;                                // 528*64*2 = 67584 B
  float*    cvec = (float*)((char*)d_ws + (size_t)WROWS * DD * 2); // 512 floats, 32B-aligned

  cc_prep<<<dim3(1), dim3(256), 0, stream>>>(NA, Nmu, U, W, cvec);
  cc_main<<<dim3(NCOLS / (32 * CTILES)), dim3(256), 0, stream>>>(XPi, U, alpha, W, cvec, out);
}
